// NeuronGatedDeltaNet_59365037965824
// MI455X (gfx1250) — matrix operand order repaired and checked by reading, not compiled, not run on hardware
//
#include <hip/hip_runtime.h>
#include <math.h>

#define BB 2
#define HH 48
#define SS 2048
#define DKK 128
#define DVV 128
#define CHUNK 64
#define DVB 32
#define NVB (DVV / DVB)       // 4 value-blocks -> 384 workgroups
#define NCHUNK (SS / CHUNK)   // 32 sequential chunks
#define NTHREADS 256
#define LDA (CHUNK + 2)       // 66: even (8B-aligned pairs), conflict-free column walks

typedef __attribute__((ext_vector_type(16))) _Float16 v16h;
typedef __attribute__((ext_vector_type(2)))  _Float16 v2h;
typedef __attribute__((ext_vector_type(8)))  float    v8f;
typedef __attribute__((ext_vector_type(2)))  float    v2f;

// ---- WMMA fragment builders (CDNA5 wave32 layouts) -------------------------
// f16 A-operand (16x32), [m][k] row-major source
__device__ __forceinline__ v16h load_a16(const _Float16* buf, int ld, int m0, int k0, int lane) {
  const int m  = m0 + (lane & 15);
  const int hs = lane >> 4;
  const _Float16* r = buf + m * ld + k0;
  v16h a;
#pragma unroll
  for (int v = 0; v < 4; ++v) {
    v2h p = *(const v2h*)(r + hs * 16 + 2 * v);
    v2h q = *(const v2h*)(r + hs * 16 + 8 + 2 * v);
    a[2 * v] = p.x; a[2 * v + 1] = p.y;
    a[8 + 2 * v] = q.x; a[9 + 2 * v] = q.y;
  }
  return a;
}
// f16 B-operand (32x16), source stored [n][k] row-major
__device__ __forceinline__ v16h load_b16(const _Float16* buf, int ld, int n0, int k0, int lane) {
  const int n = n0 + (lane & 15);
  const _Float16* r = buf + n * ld + k0 + (lane >> 4) * 16;
  v16h b;
#pragma unroll
  for (int v = 0; v < 8; ++v) {
    v2h p = *(const v2h*)(r + 2 * v);
    b[2 * v] = p.x; b[2 * v + 1] = p.y;
  }
  return b;
}
// f16 B-operand built on the fly from an f32 [n][k] buffer
__device__ __forceinline__ v16h load_b16_f32(const float* buf, int ld, int n0, int k0, int lane) {
  const int n = n0 + (lane & 15);
  const float* r = buf + n * ld + k0 + (lane >> 4) * 16;
  v16h b;
#pragma unroll
  for (int v = 0; v < 8; ++v) {
    v2f p = *(const v2f*)(r + 2 * v);
    b[2 * v] = (_Float16)p.x; b[2 * v + 1] = (_Float16)p.y;
  }
  return b;
}
// f16 A-operand built on the fly from an f32 [m][k] buffer
__device__ __forceinline__ v16h load_a16_f32(const float* buf, int ld, int m0, int k0, int lane) {
  const int m  = m0 + (lane & 15);
  const int hs = lane >> 4;
  const float* r = buf + m * ld + k0;
  v16h a;
#pragma unroll
  for (int v = 0; v < 4; ++v) {
    v2f p = *(const v2f*)(r + hs * 16 + 2 * v);
    v2f q = *(const v2f*)(r + hs * 16 + 8 + 2 * v);
    a[2 * v] = (_Float16)p.x; a[2 * v + 1] = (_Float16)p.y;
    a[8 + 2 * v] = (_Float16)q.x; a[9 + 2 * v] = (_Float16)q.y;
  }
  return a;
}

__global__ __launch_bounds__(NTHREADS)
void gdn_chunked_kernel(const float* __restrict__ Q, const float* __restrict__ K,
                        const float* __restrict__ V, const float* __restrict__ G,
                        const float* __restrict__ Beta, float* __restrict__ Out) {
  __shared__ _Float16 sK[CHUNK][DKK];       // normalized k (f16)                 16 KB
  __shared__ _Float16 sQ[CHUNK][DKK];       // normalized+scaled q (f16)          16 KB
  __shared__ float    sS0T[DVB][DKK];       // state, transposed [dv][dk], f32    16 KB
  __shared__ float    sAP[CHUNK][LDA];      // -A (solve) then P, f32             16.5 KB
  __shared__ float    sDT[DVB][LDA];        // RHS -> Delta, transposed [dv][t]    8.3 KB
  __shared__ float    sb[CHUNK], sg[CHUNK], sexpb[CHUNK], sexpCb[CHUNK], sbeta[CHUNK];
  __shared__ float    snorm[CHUNK][4];

  const int tid  = threadIdx.x;
  const int lane = tid & 31;
  const int wave = tid >> 5;
  const int hs   = lane >> 4;
  const int l16  = lane & 15;

  const int bid = blockIdx.x;
  const int vb  = bid % NVB;                // which DV block
  const int bh  = bid / NVB;                // fused (b*H + h)

  const float* qp = Q + (size_t)bh * SS * DKK;
  const float* kp = K + (size_t)bh * SS * DKK;
  const float* vp = V + (size_t)bh * SS * DVV + vb * DVB;
  const float* gp = G + (size_t)bh * SS;
  const float* bp = Beta + (size_t)bh * SS;
  float* op = Out + (size_t)bh * SS * DVV + vb * DVB;
  float* sp = Out + (size_t)BB * HH * SS * DVV + (size_t)bh * DKK * DVV + vb * DVB;

  // zero the running state
  for (int i = tid; i < DVB * DKK; i += NTHREADS) ((float*)sS0T)[i] = 0.f;
  __syncthreads();

  const int row = tid >> 2;                 // 64 rows, 4 threads per row
  const int sub = tid & 3;                  // 32 floats per thread
  const float qscale = 0.08838834764831845f;  // 1/sqrt(128)

  for (int ch = 0; ch < NCHUNK; ++ch) {
    const int s0 = ch * CHUNK;

    // ---------------- load + L2-normalize K ----------------
    {
      const float* src = kp + (size_t)(s0 + row) * DKK + sub * 32;
      float r[32]; float ss = 0.f;
#pragma unroll
      for (int j = 0; j < 8; ++j) {
        float4 f = ((const float4*)src)[j];
        r[4 * j] = f.x; r[4 * j + 1] = f.y; r[4 * j + 2] = f.z; r[4 * j + 3] = f.w;
        ss += f.x * f.x + f.y * f.y + f.z * f.z + f.w * f.w;
      }
      snorm[row][sub] = ss;
      __syncthreads();
      float tot = snorm[row][0] + snorm[row][1] + snorm[row][2] + snorm[row][3];
      const float inv = 1.f / fmaxf(sqrtf(tot), 1e-6f);
#pragma unroll
      for (int j = 0; j < 32; ++j) sK[row][sub * 32 + j] = (_Float16)(r[j] * inv);
      __syncthreads();   // snorm reused below
      // ---------------- load + L2-normalize + scale Q ----------------
      const float* srcq = qp + (size_t)(s0 + row) * DKK + sub * 32;
      ss = 0.f;
#pragma unroll
      for (int j = 0; j < 8; ++j) {
        float4 f = ((const float4*)srcq)[j];
        r[4 * j] = f.x; r[4 * j + 1] = f.y; r[4 * j + 2] = f.z; r[4 * j + 3] = f.w;
        ss += f.x * f.x + f.y * f.y + f.z * f.z + f.w * f.w;
      }
      snorm[row][sub] = ss;
      __syncthreads();
      tot = snorm[row][0] + snorm[row][1] + snorm[row][2] + snorm[row][3];
      const float invq = qscale / fmaxf(sqrtf(tot), 1e-6f);
#pragma unroll
      for (int j = 0; j < 32; ++j) sQ[row][sub * 32 + j] = (_Float16)(r[j] * invq);
    }
    // ---------------- gates: cumulative log-decay ----------------
    if (tid < CHUNK) { sg[tid] = gp[s0 + tid]; sbeta[tid] = bp[s0 + tid]; }
    __syncthreads();
    if (tid == 0) {
      float acc = 0.f;
      for (int t = 0; t < CHUNK; ++t) { acc += sg[t]; sb[t] = acc; }
    }
    __syncthreads();
    if (tid < CHUNK) {
      sexpb[tid]  = __expf(sb[tid]);                 // decay from chunk start
      sexpCb[tid] = __expf(sb[CHUNK - 1] - sb[tid]); // decay to chunk end
    }
    if (ch + 1 < NCHUNK) {                 // global_prefetch_b8 for next chunk
      __builtin_prefetch(kp + (size_t)(s0 + CHUNK + row) * DKK + sub * 32, 0, 1);
      __builtin_prefetch(qp + (size_t)(s0 + CHUNK + row) * DKK + sub * 32, 0, 1);
    }
    __syncthreads();

    // ------- Phase A: 24 tiles over 8 waves -------
    //   ids 0..15 : -A = -beta*exp(b_t-b_i)*(K K^T), strict lower (4x4 tile grid)
    //   ids 16..23: RHS = beta*(V - exp(b)*K@S0), written transposed (4x2 tile grid)
#pragma unroll
    for (int rp = 0; rp < 3; ++rp) {
      const int id = wave * 3 + rp;
      if (id < 16) {
        const int mt = id >> 2, nt = id & 3;
        v8f c = {0.f, 0.f, 0.f, 0.f, 0.f, 0.f, 0.f, 0.f};
#pragma unroll
        for (int ks = 0; ks < 4; ++ks) {
          v16h a = load_a16(&sK[0][0], DKK, mt * 16, ks * 32, lane);
          v16h b = load_b16(&sK[0][0], DKK, nt * 16, ks * 32, lane);
          c = __builtin_amdgcn_wmma_f32_16x16x32_f16(false, a, false, b, (short)0, c, false, false);
        }
        const int i = nt * 16 + l16;
#pragma unroll
        for (int v = 0; v < 8; ++v) {
          const int t = mt * 16 + hs * 8 + v;
          sAP[t][i] = (i < t) ? (-sbeta[t] * __expf(sb[t] - sb[i]) * c[v]) : 0.f;
        }
      } else {
        const int j = id - 16, mt = j >> 1, nt = j & 1;
        v8f c = {0.f, 0.f, 0.f, 0.f, 0.f, 0.f, 0.f, 0.f};
#pragma unroll
        for (int ks = 0; ks < 4; ++ks) {
          v16h a = load_a16(&sK[0][0], DKK, mt * 16, ks * 32, lane);
          v16h b = load_b16_f32(&sS0T[0][0], DKK, nt * 16, ks * 32, lane);
          c = __builtin_amdgcn_wmma_f32_16x16x32_f16(false, a, false, b, (short)0, c, false, false);
        }
        const int dv = nt * 16 + l16;
#pragma unroll
        for (int v = 0; v < 8; ++v) {
          const int t = mt * 16 + hs * 8 + v;
          const float vv = vp[(size_t)(s0 + t) * DVV + dv];  // single-use: straight from global
          sDT[dv][t] = sbeta[t] * (vv - sexpb[t] * c[v]);
        }
      }
    }
    __syncthreads();

    // ------- blocked triangular solve (I+A)Delta = RHS, 4 blocks of 16 -------
#pragma unroll
    for (int blk = 0; blk < 4; ++blk) {
      if (tid < DVB) {                     // scalar forward substitution in-block
        float loc[16];
#pragma unroll
        for (int t = 0; t < 16; ++t) {
          float acc = sDT[tid][blk * 16 + t];
          for (int i = 0; i < t; ++i) acc += sAP[blk * 16 + t][blk * 16 + i] * loc[i];  // sAP = -A
          loc[t] = acc; sDT[tid][blk * 16 + t] = acc;
        }
      }
      __syncthreads();
      if (blk < 3) {                       // right-looking update of remaining blocks (f32 WMMA k=4)
        const int ntiles = (3 - blk) * 2;
        if (wave < ntiles) {
          const int tb = blk + 1 + (wave >> 1);   // target block row
          const int n  = (wave & 1) * 16 + l16;   // dv column
          v8f c;
#pragma unroll
          for (int v = 0; v < 8; ++v) c[v] = sDT[n][tb * 16 + hs * 8 + v];
#pragma unroll
          for (int ks = 0; ks < 4; ++ks) {
            v2f a2 = *(const v2f*)&sAP[tb * 16 + l16][blk * 16 + ks * 4 + hs * 2];
            v2f b2 = *(const v2f*)&sDT[n][blk * 16 + ks * 4 + hs * 2];
            c = __builtin_amdgcn_wmma_f32_16x16x4_f32(false, a2, false, b2, (short)0, c, false, false);
          }
#pragma unroll
          for (int v = 0; v < 8; ++v) sDT[n][tb * 16 + hs * 8 + v] = c[v];
        }
        __syncthreads();
      }
    }

    // ------- Phase B: 24 tiles over 8 waves -------
    //   ids 0..15 : P = mask(Q K^T) * exp(b_t - b_i)  (diag included)
    //   ids 16..23: oc = exp(b_t) * (Q @ S0)  (kept in registers)
    v8f oc[3];
    int qmt[3], qnt[3], isq[3];
#pragma unroll
    for (int rp = 0; rp < 3; ++rp) {
      const int id = wave * 3 + rp;
      isq[rp] = 0; qmt[rp] = 0; qnt[rp] = 0;
      oc[rp] = (v8f){0.f, 0.f, 0.f, 0.f, 0.f, 0.f, 0.f, 0.f};
      if (id < 16) {
        const int mt = id >> 2, nt = id & 3;
        v8f c = {0.f, 0.f, 0.f, 0.f, 0.f, 0.f, 0.f, 0.f};
#pragma unroll
        for (int ks = 0; ks < 4; ++ks) {
          v16h a = load_a16(&sQ[0][0], DKK, mt * 16, ks * 32, lane);
          v16h b = load_b16(&sK[0][0], DKK, nt * 16, ks * 32, lane);
          c = __builtin_amdgcn_wmma_f32_16x16x32_f16(false, a, false, b, (short)0, c, false, false);
        }
        const int i = nt * 16 + l16;
#pragma unroll
        for (int v = 0; v < 8; ++v) {
          const int t = mt * 16 + hs * 8 + v;
          sAP[t][i] = (i <= t) ? (c[v] * __expf(sb[t] - sb[i])) : 0.f;
        }
      } else {
        const int j = id - 16, mt = j >> 1, nt = j & 1;
        isq[rp] = 1; qmt[rp] = mt; qnt[rp] = nt;
        v8f c = {0.f, 0.f, 0.f, 0.f, 0.f, 0.f, 0.f, 0.f};
#pragma unroll
        for (int ks = 0; ks < 4; ++ks) {
          v16h a = load_a16(&sQ[0][0], DKK, mt * 16, ks * 32, lane);
          v16h b = load_b16_f32(&sS0T[0][0], DKK, nt * 16, ks * 32, lane);
          c = __builtin_amdgcn_wmma_f32_16x16x32_f16(false, a, false, b, (short)0, c, false, false);
        }
#pragma unroll
        for (int v = 0; v < 8; ++v) { const int t = mt * 16 + hs * 8 + v; c[v] *= sexpb[t]; }
        oc[rp] = c;
      }
    }
    __syncthreads();
    // ------- Phase B2: O = oc + P @ Delta, store to global -------
#pragma unroll
    for (int rp = 0; rp < 3; ++rp) {
      if (isq[rp]) {
        const int mt = qmt[rp], nt = qnt[rp];
        v8f c = oc[rp];
#pragma unroll
        for (int ks = 0; ks < 2; ++ks) {
          v16h a = load_a16_f32(&sAP[0][0], LDA, mt * 16, ks * 32, lane);
          v16h b = load_b16_f32(&sDT[0][0], LDA, nt * 16, ks * 32, lane);
          c = __builtin_amdgcn_wmma_f32_16x16x32_f16(false, a, false, b, (short)0, c, false, false);
        }
        const int dv = nt * 16 + l16;
#pragma unroll
        for (int v = 0; v < 8; ++v) {
          const int t = mt * 16 + hs * 8 + v;
          op[(size_t)(s0 + t) * DVV + dv] = c[v];
        }
      }
    }
    __syncthreads();

    // ------- M5 (all waves): S0 = exp(b_C)*S0 + sum_t exp(b_C-b_t) k_t delta_t^T -------
    const float ebC = sexpb[CHUNK - 1];
#pragma unroll
    for (int rep = 0; rep < 2; ++rep) {
      const int tt = wave * 2 + rep;       // 16 tiles over 8 waves
      const int mt = tt & 1;               // dv block
      const int nt = tt >> 1;              // dk block 0..7
      const int n  = nt * 16 + l16;        // dk index
      v8f c;
#pragma unroll
      for (int v = 0; v < 8; ++v) c[v] = ebC * sS0T[mt * 16 + hs * 8 + v][n];
#pragma unroll
      for (int ks = 0; ks < 2; ++ks) {
        v16h a = load_a16_f32(&sDT[0][0], LDA, mt * 16, ks * 32, lane);  // Delta^T [dv][t]
        v16h b;
#pragma unroll
        for (int v = 0; v < 8; ++v) {      // K with end-of-chunk decay, layout [t][dk]
          const int t = ks * 32 + hs * 16 + 2 * v;
          b[2 * v]     = (_Float16)((float)sK[t][n] * sexpCb[t]);
          b[2 * v + 1] = (_Float16)((float)sK[t + 1][n] * sexpCb[t + 1]);
        }
        c = __builtin_amdgcn_wmma_f32_16x16x32_f16(false, a, false, b, (short)0, c, false, false);
      }
#pragma unroll
      for (int v = 0; v < 8; ++v) sS0T[mt * 16 + hs * 8 + v][n] = c[v];
    }
    __syncthreads();
  }

  // ---------------- write final state [B,H,DK,DV] ----------------
  for (int i = tid; i < DVB * DKK; i += NTHREADS) {
    const int dv = i / DKK, dk = i % DKK;
    sp[(size_t)dk * DVV + dv] = sS0T[dv][dk];
  }
}

extern "C" void kernel_launch(void* const* d_in, const int* in_sizes, int n_in,
                              void* d_out, int out_size, void* d_ws, size_t ws_size,
                              hipStream_t stream) {
  (void)in_sizes; (void)n_in; (void)d_ws; (void)ws_size; (void)out_size;
  const float* q    = (const float*)d_in[0];
  const float* k    = (const float*)d_in[1];
  const float* v    = (const float*)d_in[2];
  const float* g    = (const float*)d_in[3];
  const float* beta = (const float*)d_in[4];
  float* out = (float*)d_out;
  dim3 grid(BB * HH * NVB);
  gdn_chunked_kernel<<<grid, NTHREADS, 0, stream>>>(q, k, v, g, beta, out);
}
